// DGCNCritic_62079457296721
// MI455X (gfx1250) — matrix operand order repaired and checked by reading, not compiled, not run on hardware
//
#include <hip/hip_runtime.h>

// ---------------- CDNA5 WMMA GEMM infrastructure ----------------
typedef __attribute__((ext_vector_type(16))) _Float16 v16h;
typedef __attribute__((ext_vector_type(8)))  _Float16 v8h;
typedef __attribute__((ext_vector_type(8)))  float    v8f;

#define B_    4096
#define A_    8
#define D_    256
#define H_    256
#define NL_   4
#define FC_   512
#define FOURH 1024
#define MAX_SRC 10

struct GemmSrc {
  const void*     X;       // f32 or f16 activations, row r at X + r*ldx (elements)
  long long       ldx;
  int             xIsF16;  // 1: X is _Float16 (no scale); 0: X is float (optional scale)
  int             K;       // reduction length (multiple of 32)
  const _Float16* W;       // pre-converted f16 weights, (N,K) row-major
  long long       ldw;
  const float*    scale;   // optional per-row scale (mask), scale[r*scaleStride]
  int             scaleStride;
};

struct GemmArgs {
  GemmSrc      s[MAX_SRC];
  int          nsrc;
  const float* bias;   // optional, length N
  const float* bias2;  // optional, length N
  float*       C;      // f32 output (always written)
  _Float16*    C16;    // optional f16 copy of output (same ldc)
  long long    ldc;
  int          N;
  int          act;    // 0 none, 1 relu
};

// One wave computes a 16(M) x 64(N) strip of C: one A fragment feeds 4
// v_wmma_f32_16x16x32_f16 per K-step.  Block = 128 threads = 4 waves stacked
// on M -> 64x64 macro-tile.  All weights are pre-converted f16 (N,K) so the
// B fragment is two 16B vector loads (k and k+16), no VALU converts.
__global__ __launch_bounds__(128)
void gemm_wmma(GemmArgs g) {
  const int lane  = threadIdx.x & 31;
  const int wave  = threadIdx.x >> 5;
  const int mtile = (blockIdx.x << 6) + (wave << 4);
  const int ntile = blockIdx.y << 6;

  const int mrow = mtile + (lane & 15);      // A-frag: lane carries row M = lane%16
  const int nlow = lane & 15;                // B-frag: lane carries col N = lane%16
  const int aK   = (lane >> 4) << 3;         // A layout: lanes 0-15 K in [0,8)+[16,24); lanes 16-31 +8
  const int bK   = (lane >> 4) << 3;         // B layout: same as A (lanes 0-15 K in [0,8)+[16,24); lanes 16-31 +8)

  v8f acc[4];
#pragma unroll
  for (int j = 0; j < 4; ++j) acc[j] = v8f{0.f,0.f,0.f,0.f,0.f,0.f,0.f,0.f};

  for (int si = 0; si < g.nsrc; ++si) {
    const GemmSrc s = g.s[si];
    const _Float16* Wn[4];
#pragma unroll
    for (int j = 0; j < 4; ++j)
      Wn[j] = s.W + (long long)(ntile + (j << 4) + nlow) * s.ldw;

    if (s.xIsF16) {
      const _Float16* Xr = (const _Float16*)s.X + (long long)mrow * s.ldx;
      for (int kk = 0; kk < s.K; kk += 32) {
        v16h af;
        v8h a0 = *(const v8h*)(Xr + kk + aK);
        v8h a1 = *(const v8h*)(Xr + kk + aK + 16);
#pragma unroll
        for (int u = 0; u < 8; ++u) { af[u] = a0[u]; af[u + 8] = a1[u]; }
#pragma unroll
        for (int j = 0; j < 4; ++j) {
          v16h bf = __builtin_shufflevector(*(const v8h*)(Wn[j] + kk + bK), *(const v8h*)(Wn[j] + kk + bK + 16), 0,1,2,3,4,5,6,7,8,9,10,11,12,13,14,15);
          acc[j] = __builtin_amdgcn_wmma_f32_16x16x32_f16(false, af, false, bf,
                                                          (short)0, acc[j], false, false);
        }
      }
    } else {
      const float* Xr = (const float*)s.X + (long long)mrow * s.ldx;
      const float  sc = s.scale ? s.scale[(long long)mrow * s.scaleStride] : 1.0f;
      for (int kk = 0; kk < s.K; kk += 32) {
        v16h af;
#pragma unroll
        for (int u = 0; u < 8; ++u) {
          af[u]     = (_Float16)(Xr[kk + aK + u]      * sc);
          af[u + 8] = (_Float16)(Xr[kk + aK + 16 + u] * sc);
        }
#pragma unroll
        for (int j = 0; j < 4; ++j) {
          v16h bf = __builtin_shufflevector(*(const v8h*)(Wn[j] + kk + bK), *(const v8h*)(Wn[j] + kk + bK + 16), 0,1,2,3,4,5,6,7,8,9,10,11,12,13,14,15);
          acc[j] = __builtin_amdgcn_wmma_f32_16x16x32_f16(false, af, false, bf,
                                                          (short)0, acc[j], false, false);
        }
      }
    }
  }

  const int mbase = mtile + ((lane >> 4) << 3);  // C layout: VGPR r -> M = r + (lane<16?0:8)
#pragma unroll
  for (int j = 0; j < 4; ++j) {
    const int ncol = ntile + (j << 4) + nlow;
    float bn = 0.f;
    if (g.bias)  bn += g.bias[ncol];
    if (g.bias2) bn += g.bias2[ncol];
#pragma unroll
    for (int r = 0; r < 8; ++r) {
      float v = acc[j][r] + bn;
      if (g.act) v = fmaxf(v, 0.f);
      const long long o = (long long)(mbase + r) * g.ldc + ncol;
      g.C[o] = v;
      if (g.C16) g.C16[o] = (_Float16)v;
    }
  }
}

// ---------------- weight staging kernels (one pass per weight per launch) ----

// plain f32 -> f16, (N,K) layout preserved; 2 elements / thread
__global__ void cvt16_k(const float* __restrict__ s, _Float16* __restrict__ d) {
  long long i = ((long long)blockIdx.x * blockDim.x + threadIdx.x) * 2;
  d[i]     = (_Float16)s[i];
  d[i + 1] = (_Float16)s[i + 1];
}

// transpose + convert: s is (K,N) f32 -> d is (N,K) f16
__global__ void cvtT16_k(const float* __restrict__ s, _Float16* __restrict__ d,
                         int K, int N) {
  long long i = (long long)blockIdx.x * blockDim.x + threadIdx.x;  // over N*K
  int n = (int)(i / K), k = (int)(i % K);
  d[i] = (_Float16)s[(long long)k * N + n];
}

// ---------------- element-wise kernels ----------------

// m[b,d] = mean_a obs[b,a,d]  (f16 copy for WMMA consumption)
__global__ void agent_mean_k(const float* __restrict__ obs, float* __restrict__ m,
                             _Float16* __restrict__ m16) {
  int idx = blockIdx.x * blockDim.x + threadIdx.x;  // B*D threads
  int b = idx >> 8, d = idx & 255;
  float s = 0.f;
#pragma unroll
  for (int a = 0; a < A_; ++a) s += obs[((long long)b * A_ + a) * D_ + d];
  float v = s * 0.125f;
  m[idx] = v;
  m16[idx] = (_Float16)v;
}

__device__ __forceinline__ float sigm(float x) { return 1.f / (1.f + __expf(-x)); }

// gates (B,1024) -> h,c.  c0/hOut/cOut pre-offset by (a*NL+l)*H, row stride A*NL*H.
// mask pre-offset by a, stride A.  h16: contiguous (B,256) f16 copy of h.
__global__ void lstm_cell_k(const float* __restrict__ gates, const float* __restrict__ c0,
                            const float* __restrict__ mask,
                            float* __restrict__ hOut, float* __restrict__ cOut,
                            _Float16* __restrict__ h16) {
  int idx = blockIdx.x * blockDim.x + threadIdx.x;  // B*H threads
  int b = idx >> 8, j = idx & 255;
  const float* gr = gates + (long long)b * FOURH;
  float gi = gr[j], gf = gr[H_ + j], gg = gr[2 * H_ + j], go = gr[3 * H_ + j];
  float mk = mask[(long long)b * A_];
  float cp = c0[(long long)b * (A_ * NL_ * H_) + j] * mk;
  float c  = sigm(gf) * cp + sigm(gi) * tanhf(gg);
  float h  = sigm(go) * tanhf(c);
  hOut[(long long)b * (A_ * NL_ * H_) + j] = h;
  cOut[(long long)b * (A_ * NL_ * H_) + j] = c;
  h16[idx] = (_Float16)h;
}

// v[b*A + a] = dot(h[b,:], w) + vb ; one wave per row
__global__ void vhead_k(const float* __restrict__ h, const float* __restrict__ w,
                        const float* __restrict__ vb, float* __restrict__ out, int a) {
  int row  = blockIdx.x * 8 + (threadIdx.x >> 5);
  int lane = threadIdx.x & 31;
  const float* hr = h + (long long)row * FC_;
  float s = 0.f;
  for (int k = lane; k < FC_; k += 32) s += hr[k] * w[k];
#pragma unroll
  for (int off = 16; off > 0; off >>= 1) s += __shfl_xor(s, off, 32);
  if (lane == 0) out[(long long)row * A_ + a] = s + vb[0];
}

// ---------------- host helpers ----------------
static inline GemmSrc srcF32(const float* X, long long ldx, int K,
                             const _Float16* W, long long ldw,
                             const float* scale = nullptr, int ss = 0) {
  GemmSrc s; s.X = X; s.ldx = ldx; s.xIsF16 = 0; s.K = K;
  s.W = W; s.ldw = ldw; s.scale = scale; s.scaleStride = ss; return s;
}
static inline GemmSrc srcF16(const _Float16* X, long long ldx, int K,
                             const _Float16* W, long long ldw) {
  GemmSrc s; s.X = X; s.ldx = ldx; s.xIsF16 = 1; s.K = K;
  s.W = W; s.ldw = ldw; s.scale = nullptr; s.scaleStride = 0; return s;
}
static inline void run_gemm(hipStream_t st, const GemmArgs& g, int M) {
  dim3 grid(M / 64, g.N / 64);
  gemm_wmma<<<grid, 128, 0, st>>>(g);
}
static inline void stage_cvt(hipStream_t st, const float* s, _Float16* d, long long n) {
  cvt16_k<<<(unsigned)(n / 512), 256, 0, st>>>(s, d);
}
static inline void stage_cvtT(hipStream_t st, const float* s, _Float16* d, int K, int N) {
  cvtT16_k<<<(unsigned)(((long long)K * N) / 256), 256, 0, st>>>(s, d, K, N);
}

extern "C" void kernel_launch(void* const* d_in, const int* in_sizes, int n_in,
                              void* d_out, int out_size, void* d_ws, size_t ws_size,
                              hipStream_t stream) {
  (void)in_sizes; (void)n_in; (void)out_size; (void)ws_size;
  const float* obs   = (const float*)d_in[0];
  const float* s_h   = (const float*)d_in[1];
  const float* s_c   = (const float*)d_in[2];
  const float* g_h   = (const float*)d_in[3];
  const float* g_c   = (const float*)d_in[4];
  const float* masks = (const float*)d_in[5];
  const float* dg_w  = (const float*)d_in[6];
  const float* dg_b  = (const float*)d_in[7];
  const float* s_wih = (const float*)d_in[8];
  const float* s_whh = (const float*)d_in[9];
  const float* s_bih = (const float*)d_in[10];
  const float* s_bhh = (const float*)d_in[11];
  const float* c_wih = (const float*)d_in[12];
  const float* c_whh = (const float*)d_in[13];
  const float* c_bih = (const float*)d_in[14];
  const float* c_bhh = (const float*)d_in[15];
  const float* a_wi  = (const float*)d_in[16];
  const float* a_bi  = (const float*)d_in[17];
  const float* a_wo  = (const float*)d_in[18];
  const float* a_bo  = (const float*)d_in[19];
  const float* w0    = (const float*)d_in[20];
  const float* b0    = (const float*)d_in[21];
  const float* w1    = (const float*)d_in[22];
  const float* b1    = (const float*)d_in[23];
  const float* vw    = (const float*)d_in[24];
  const float* vb    = (const float*)d_in[25];

  float* out = (float*)d_out;
  const long long BA  = (long long)B_ * A_;
  const long long blk = (long long)B_ * A_ * NL_ * H_;
  float* outV = out;
  float* nshO = out + BA;
  float* nscO = nshO + blk;
  float* nghO = nscO + blk;
  float* ngcO = nghO + blk;

  // f32 workspace
  float* ws   = (float*)d_ws;
  float* wm   = ws;
  float* wh1  = wm   + (long long)B_ * 256;
  float* wdg  = wh1  + (long long)B_ * 256;
  float* wg   = wdg  + (long long)B_ * 256;   // gates B*1024
  float* wvp  = wg   + (long long)B_ * 1024;
  float* wats = wvp  + (long long)B_ * 256;
  float* watg = wats + (long long)B_ * 256;
  float* wf0  = watg + (long long)B_ * 256;   // B*512
  float* wf1  = wf0  + (long long)B_ * 512;   // B*512
  // f16 workspace (after B_*3584 floats; 32B aligned by construction)
  _Float16* h16base = (_Float16*)(wf1 + (long long)B_ * 512);
  _Float16* m16   = h16base;                         // B*256 (agent mean)
  _Float16* h116  = m16   + (long long)B_ * 256;     // dgcn layer-1 out
  _Float16* dg16  = h116  + (long long)B_ * 256;     // dgcn out
  _Float16* h16   = dg16  + (long long)B_ * 256;     // LSTM layer h (reused per layer)
  _Float16* vp16  = h16   + (long long)B_ * 256;     // attention v-proj
  _Float16* ats16 = vp16  + (long long)B_ * 256;     // att_s vector
  _Float16* atg16 = ats16 + (long long)B_ * 256;     // att_g vector
  _Float16* f016  = atg16 + (long long)B_ * 256;     // fc0 out, B*512
  _Float16* wstg  = f016  + (long long)B_ * 512;     // weight staging (<= 2560*512 halfs)

  // ---- DGCN: mean over agents, then two (B,256)x(256,256) relu GEMMs.
  // (mean of broadcast == identity, so layer 2 feeds on layer-1 output directly)
  agent_mean_k<<<(B_ * D_) / 256, 256, 0, stream>>>(obs, wm, m16);
  {
    GemmArgs g = {};
    g.nsrc = 1;
    stage_cvtT(stream, dg_w, wstg, 256, 256);            // (K,N) -> (N,K) f16
    g.s[0] = srcF16(m16, 256, 256, wstg, 256);
    g.bias = dg_b; g.C = wh1; g.C16 = h116; g.ldc = 256; g.N = 256; g.act = 1;
    run_gemm(stream, g, B_);
    stage_cvtT(stream, dg_w + 256 * 256, wstg, 256, 256);
    g.s[0] = srcF16(h116, 256, 256, wstg, 256);
    g.bias = dg_b + 256; g.C = wdg; g.C16 = dg16;
    run_gemm(stream, g, B_);
  }

  auto run_lstm = [&](const float* x0, long long x0ld,
                      const float* wih, const float* whh,
                      const float* bih, const float* bhh,
                      const float* h0, const float* c0,
                      float* hOut, float* cOut, int a, const _Float16* x0f16) {
    for (int l = 0; l < NL_; ++l) {
      const long long po = (long long)a * NL_ + l;
      _Float16* wihS = wstg;                 // 1024*256 halfs
      _Float16* whhS = wstg + FOURH * 256;   // 1024*256 halfs
      stage_cvt(stream, wih + po * FOURH * D_, wihS, (long long)FOURH * D_);
      stage_cvt(stream, whh + po * FOURH * H_, whhS, (long long)FOURH * H_);
      GemmArgs g = {};
      g.nsrc = 2;
      if (l == 0) {
        if (x0f16) g.s[0] = srcF16(x0f16, x0ld, 256, wihS, D_);
        else       g.s[0] = srcF32(x0,    x0ld, 256, wihS, D_);
      } else {
        g.s[0] = srcF16(h16, 256, 256, wihS, D_);   // previous layer h (f16 copy)
      }
      g.s[1] = srcF32(h0 + po * H_, A_ * NL_ * H_, 256, whhS, H_, masks + a, A_);
      g.bias = bih + po * FOURH; g.bias2 = bhh + po * FOURH;
      g.C = wg; g.C16 = nullptr; g.ldc = FOURH; g.N = FOURH; g.act = 0;
      run_gemm(stream, g, B_);
      lstm_cell_k<<<(B_ * H_) / 256, 256, 0, stream>>>(wg, c0 + po * H_, masks + a,
                                                       hOut + po * H_, cOut + po * H_, h16);
    }
  };

  // MHA with kv-seq-len 1: softmax over 1 key == 1, so output = (so@Wv^T+bv)@Wo^T+bo,
  // identical for every query row (handled as 4 repeated W-segments in fc0).
  // Uses h16 (= last LSTM layer h, f16) as input -> run immediately after its LSTM.
  auto run_att = [&](float* attF32, _Float16* attF16, int a) {
    stage_cvt(stream, a_wi + ((long long)a * 3 * H_ + 2 * H_) * H_, wstg, (long long)H_ * H_);
    GemmArgs g = {};
    g.nsrc = 1;
    g.s[0] = srcF16(h16, 256, 256, wstg, H_);
    g.bias = a_bi + (long long)a * 3 * H_ + 2 * H_;
    g.C = wvp; g.C16 = vp16; g.ldc = 256; g.N = 256; g.act = 0;
    run_gemm(stream, g, B_);
    stage_cvt(stream, a_wo + (long long)a * H_ * H_, wstg, (long long)H_ * H_);
    GemmArgs g2 = {};
    g2.nsrc = 1;
    g2.s[0] = srcF16(vp16, 256, 256, wstg, H_);
    g2.bias = a_bo + (long long)a * H_;
    g2.C = attF32; g2.C16 = attF16; g2.ldc = 256; g2.N = 256; g2.act = 0;
    run_gemm(stream, g2, B_);
  };

  for (int a = 0; a < A_; ++a) {
    // somu LSTM on obs -> att_s (needs h16 of somu layer 3)
    run_lstm(obs + (long long)a * D_, (long long)A_ * D_,
             s_wih, s_whh, s_bih, s_bhh, s_h, s_c, nshO, nscO, a, nullptr);
    run_att(wats, ats16, a);
    // scmu LSTM on dgcn output (broadcast -> shared dg16) -> att_g
    run_lstm(nullptr, 256,
             c_wih, c_whh, c_bih, c_bhh, g_h, g_c, nghO, ngcO, a, dg16);
    run_att(watg, atg16, a);

    // fc0: feat = [obs_a(256) | dg(256) | att_s x4 | att_g x4] -> 10 K-segments of 256
    {
      stage_cvtT(stream, w0 + (long long)a * 2560 * FC_, wstg, 2560, FC_); // -> (512,2560) f16
      GemmArgs g = {};
      g.nsrc = 10;
      g.s[0] = srcF32(obs + (long long)a * D_, (long long)A_ * D_, 256, wstg, 2560);
      g.s[1] = srcF16(dg16, 256, 256, wstg + 256, 2560);
      for (int q = 0; q < 4; ++q) {
        g.s[2 + q] = srcF16(ats16, 256, 256, wstg + 512  + q * 256, 2560);
        g.s[6 + q] = srcF16(atg16, 256, 256, wstg + 1536 + q * 256, 2560);
      }
      g.bias = b0 + (long long)a * FC_;
      g.C = wf0; g.C16 = f016; g.ldc = FC_; g.N = FC_; g.act = 1;
      run_gemm(stream, g, B_);
    }
    // fc1: (B,512)x(512,512) relu
    {
      stage_cvtT(stream, w1 + (long long)a * FC_ * FC_, wstg, FC_, FC_);  // -> (512,512) f16
      GemmArgs g = {};
      g.nsrc = 1;
      g.s[0] = srcF16(f016, FC_, FC_, wstg, FC_);
      g.bias = b1 + (long long)a * FC_;
      g.C = wf1; g.C16 = nullptr; g.ldc = FC_; g.N = FC_; g.act = 1;
      run_gemm(stream, g, B_);
    }
    // value head
    vhead_k<<<B_ / 8, 256, 0, stream>>>(wf1, vw, vb, outV, a);
  }
}
